// sequence_2_sequence_LSTM_15556371546700
// MI455X (gfx1250) — matrix operand order repaired and checked by reading, not compiled, not run on hardware
//
#include <hip/hip_runtime.h>
#include <math.h>

// ---------------------------------------------------------------------------
// Types for CDNA5 WMMA (wave32): v_wmma_f32_16x16x32_bf16
// ---------------------------------------------------------------------------
typedef __attribute__((ext_vector_type(16))) __bf16         v16bf;
typedef __attribute__((ext_vector_type(8)))  float          v8f;
typedef __attribute__((ext_vector_type(8)))  unsigned short u16x8;
typedef __attribute__((ext_vector_type(4)))  int            v4i;

union FragA { u16x8 h[2]; v16bf v; };

__device__ __forceinline__ unsigned short f2bf(float f) {
    union { float f; unsigned int u; } x; x.f = f;
    unsigned int r = x.u + 0x7FFFu + ((x.u >> 16) & 1u);   // round-to-nearest-even
    return (unsigned short)(r >> 16);
}

__device__ __forceinline__ float sigm(float x) { return 1.0f / (1.0f + __expf(-x)); }

// ---------------------------------------------------------------------------
// Async global->LDS copy of 16 bytes per lane (ASYNCcnt-tracked CDNA5 path).
// Builtin signature (from toolchain): (v4i AS1* src, v4i AS3* dst, imm, imm cpol)
// ---------------------------------------------------------------------------
__device__ __forceinline__ void async_copy16(const unsigned short* g, unsigned short* s)
{
#if __has_builtin(__builtin_amdgcn_global_load_async_to_lds_b128)
    __builtin_amdgcn_global_load_async_to_lds_b128(
        (__attribute__((address_space(1))) v4i*)(unsigned short*)g,
        (__attribute__((address_space(3))) v4i*)s,
        0, 0);
#else
    unsigned lds_off = (unsigned)(size_t)(__attribute__((address_space(3))) void*)s;
    asm volatile("global_load_async_to_lds_b128 %0, %1, off"
                 :: "v"(lds_off), "v"((unsigned long long)(size_t)g) : "memory");
#endif
}

__device__ __forceinline__ void wait_async0()
{
#if __has_builtin(__builtin_amdgcn_s_wait_asynccnt)
    __builtin_amdgcn_s_wait_asynccnt(0);
#else
    asm volatile("s_wait_asynccnt 0x0" ::: "memory");
#endif
}

// ---------------------------------------------------------------------------
// bf16 WMMA GEMM, LDS double-buffered:
//   C[M,N](f32) = A[M,K](bf16 row-major) * Bt[N,K]^T  (+bias | +=C)
// Block = 256 threads (8 waves), macro-tile 128(M) x 64(N), k-step 32.
// Wave grid 4x2; each wave computes 32x32 = 2x2 WMMA tiles.
// Requires: M % 128 == 0, N % 64 == 0, K % 32 == 0.
// LDS rows padded to 40 elems (80B) -> conflict-free ds_load_b128 fragments.
// ---------------------------------------------------------------------------
#define LDSPITCH 40

__global__ void __launch_bounds__(256)
gemm_bf16_wmma(const unsigned short* __restrict__ A, int lda,
               const unsigned short* __restrict__ Bt, int ldb,
               float* __restrict__ C, int ldc,
               const float* __restrict__ bias, int biasN,
               int M, int N, int K, int accumulate)
{
    __shared__ __align__(16) unsigned short Asm[2][128 * LDSPITCH];
    __shared__ __align__(16) unsigned short Bsm[2][ 64 * LDSPITCH];

    const int tid     = threadIdx.x;
    const int wave    = tid >> 5;
    const int lane    = tid & 31;
    const int lm      = lane & 15;
    const int g       = lane >> 4;

    const int blocksN = N >> 6;
    const int bM      = blockIdx.x / blocksN;
    const int bN      = blockIdx.x % blocksN;

    const int waveM   = wave & 3;          // 4 waves along M (4*32 = 128)
    const int waveN   = wave >> 2;         // 2 waves along N (2*32 = 64)

    const long mBase  = (long)bM * 128;    // global row base of block
    const long nBase  = (long)bN * 64;     // global col base of block

    // ---- panel loader: A 128x32 (2 passes) + B 64x32 (1 pass), 16B per lane ----
    const int ar  = tid >> 2;              // 0..63   (row within pass)
    const int ac8 = (tid & 3) * 8;         // k-chunk: 0,8,16,24
    auto load_panels = [&](int k0, int buf) {
        #pragma unroll
        for (int p = 0; p < 2; ++p) {
            int r = p * 64 + ar;
            async_copy16(A + (mBase + r) * (size_t)lda + k0 + ac8,
                         &Asm[buf][r * LDSPITCH + ac8]);
        }
        async_copy16(Bt + (nBase + ar) * (size_t)ldb + k0 + ac8,
                     &Bsm[buf][ar * LDSPITCH + ac8]);
    };

    // ---- accumulator init ----
    v8f acc[2][2];
    const int aRow0 = waveM * 32;          // local A-row base of this wave
    const int bCol0 = waveN * 32;          // local B-col base of this wave
    if (accumulate) {
        #pragma unroll
        for (int i = 0; i < 2; ++i)
            #pragma unroll
            for (int j = 0; j < 2; ++j) {
                long r0 = mBase + aRow0 + 16 * i + g * 8;
                long n  = nBase + bCol0 + 16 * j + lm;
                #pragma unroll
                for (int r = 0; r < 8; ++r) acc[i][j][r] = C[(r0 + r) * (size_t)ldc + n];
            }
    } else if (bias) {
        #pragma unroll
        for (int j = 0; j < 2; ++j) {
            long n = nBase + bCol0 + 16 * j + lm;
            float bv = (n < biasN) ? bias[n] : 0.0f;
            #pragma unroll
            for (int i = 0; i < 2; ++i)
                #pragma unroll
                for (int r = 0; r < 8; ++r) acc[i][j][r] = bv;
        }
    } else {
        #pragma unroll
        for (int i = 0; i < 2; ++i)
            #pragma unroll
            for (int j = 0; j < 2; ++j)
                #pragma unroll
                for (int r = 0; r < 8; ++r) acc[i][j][r] = 0.0f;
    }

    // ---- k-loop, double buffered ----
    const int ka = g * 8;                  // A fragment K-base within 32-slice
    const int kb = g * 8;                  // B fragment K-base within 32-slice

    load_panels(0, 0);
    wait_async0();
    __syncthreads();

    int cur = 0;
    for (int k0 = 0; k0 < K; k0 += 32) {
        if (k0 + 32 < K) load_panels(k0 + 32, cur ^ 1);   // overlap fill of next buffer

        FragA fa[2], fb[2];
        #pragma unroll
        for (int i = 0; i < 2; ++i) {
            const unsigned short* ap = &Asm[cur][(aRow0 + 16 * i + lm) * LDSPITCH + ka];
            fa[i].h[0] = *(const u16x8*)(ap);
            fa[i].h[1] = *(const u16x8*)(ap + 16);
        }
        #pragma unroll
        for (int j = 0; j < 2; ++j) {
            const unsigned short* bp = &Bsm[cur][(bCol0 + 16 * j + lm) * LDSPITCH + kb];
            fb[j].h[0] = *(const u16x8*)(bp);
            fb[j].h[1] = *(const u16x8*)(bp + 16);
        }

        #pragma unroll
        for (int i = 0; i < 2; ++i)
            #pragma unroll
            for (int j = 0; j < 2; ++j)
                acc[i][j] = __builtin_amdgcn_wmma_f32_16x16x32_bf16(
                                false, fa[i].v, false, fb[j].v,
                                (short)0, acc[i][j], false, false);

        wait_async0();                     // next buffer resident
        __syncthreads();                   // everyone done reading cur
        cur ^= 1;
    }

    // ---- store ----
    #pragma unroll
    for (int i = 0; i < 2; ++i)
        #pragma unroll
        for (int j = 0; j < 2; ++j) {
            long r0 = mBase + aRow0 + 16 * i + g * 8;
            long n  = nBase + bCol0 + 16 * j + lm;
            #pragma unroll
            for (int r = 0; r < 8; ++r) C[(r0 + r) * (size_t)ldc + n] = acc[i][j][r];
        }
}

// ---------------------------------------------------------------------------
// frames [B,T,D] f32  ->  frames_bf16 [T,B,D]
// ---------------------------------------------------------------------------
__global__ void conv_frames(const float* __restrict__ src, unsigned short* __restrict__ dst,
                            int B, int T, int D)
{
    long i = (long)blockIdx.x * blockDim.x + threadIdx.x;
    long total = (long)B * T * D;
    if (i >= total) return;
    int d = (int)(i % D);
    int t = (int)((i / D) % T);
    int b = (int)(i / ((long)T * D));
    dst[((size_t)t * B + b) * D + d] = f2bf(src[i]);
}

// ---------------------------------------------------------------------------
// Weight convert + transpose + pad:
// src[k+kOff, n] (srcLd cols, f32) -> out[n, k] bf16, n<Npad, k<Kpad (zero pad)
// ---------------------------------------------------------------------------
__global__ void conv_weight(const float* __restrict__ src, int srcLd, int kOff,
                            int K, int Kpad, int N, int Npad,
                            unsigned short* __restrict__ out)
{
    long i = (long)blockIdx.x * blockDim.x + threadIdx.x;
    long total = (long)Npad * Kpad;
    if (i >= total) return;
    int k = (int)(i % Kpad);
    int n = (int)(i / Kpad);
    float v = (k < K && n < N) ? src[(size_t)(kOff + k) * srcLd + n] : 0.0f;
    out[i] = f2bf(v);
}

// ---------------------------------------------------------------------------
// Embedding gather: cap_emb[t,b,d] = emb[ids[b,t], d] (d<Dw else 0), bf16 [T,B,Dpad]
// ---------------------------------------------------------------------------
__global__ void gather_emb(const int* __restrict__ ids, const float* __restrict__ table,
                           unsigned short* __restrict__ out, int T, int B, int Dw, int Dpad)
{
    long i = (long)blockIdx.x * blockDim.x + threadIdx.x;
    long total = (long)T * B * Dpad;
    if (i >= total) return;
    int d = (int)(i % Dpad);
    int b = (int)((i / Dpad) % B);
    int t = (int)(i / ((long)B * Dpad));
    float v = 0.0f;
    if (d < Dw) {
        int id = ids[(size_t)b * T + t];
        v = table[(size_t)id * Dw + d];
    }
    out[i] = f2bf(v);
}

// ---------------------------------------------------------------------------
// LSTM cell (elementwise): gates [B,4H] f32 (i,j,f,o), updates c (f32), h (bf16)
// ---------------------------------------------------------------------------
__global__ void lstm_cell(const float* __restrict__ gates, float* __restrict__ c,
                          unsigned short* __restrict__ h_bf,
                          unsigned short* __restrict__ h_store,  // nullable
                          int B, int H)
{
    int i = blockIdx.x * blockDim.x + threadIdx.x;
    if (i >= B * H) return;
    int b = i / H, hh = i % H;
    const float* g = gates + (size_t)b * 4 * H;
    float gi = g[hh], gj = g[H + hh], gf = g[2 * H + hh], go = g[3 * H + hh];
    float cn = c[i] * sigm(gf + 1.0f) + sigm(gi) * tanhf(gj);   // forget bias 1.0
    float hn = tanhf(cn) * sigm(go);
    c[i] = cn;
    unsigned short hb = f2bf(hn);
    h_bf[i] = hb;
    if (h_store) h_store[i] = hb;
}

// ---------------------------------------------------------------------------
// proj [T*B, Npad] f32  ->  d_out [B, T, Dw] f32
// ---------------------------------------------------------------------------
__global__ void write_out(const float* __restrict__ proj, float* __restrict__ out,
                          int T, int B, int Dw, int Npad)
{
    long i = (long)blockIdx.x * blockDim.x + threadIdx.x;
    long total = (long)B * T * Dw;
    if (i >= total) return;
    int w = (int)(i % Dw);
    int t = (int)((i / Dw) % T);
    int b = (int)(i / ((long)T * Dw));
    out[i] = proj[((size_t)t * B + b) * Npad + w];
}

// ---------------------------------------------------------------------------
// Orchestration
// ---------------------------------------------------------------------------
static inline unsigned gsz(long n, int b) { return (unsigned)((n + b - 1) / b); }

extern "C" void kernel_launch(void* const* d_in, const int* in_sizes, int n_in,
                              void* d_out, int out_size, void* d_ws, size_t ws_size,
                              hipStream_t stream)
{
    (void)in_sizes; (void)n_in; (void)out_size; (void)ws_size;

    constexpr int B    = 128;
    constexpr int TE   = 64;      // T_enc
    constexpr int DIN  = 2048;
    constexpr int TD   = 32;      // T_dec
    constexpr int DW   = 300;     // word dim
    constexpr int DWP  = 320;     // padded K for decoder input
    constexpr int H    = 512;
    constexpr int G4   = 4 * H;   // 2048
    constexpr int NPO  = 320;     // padded N for output projection (mult of 64)

    const float* frames  = (const float*)d_in[0];
    const int*   ids     = (const int*)  d_in[1];
    const float* emb     = (const float*)d_in[2];
    const float* W_enc   = (const float*)d_in[3];
    const float* b_enc   = (const float*)d_in[4];
    const float* W_dec   = (const float*)d_in[5];
    const float* b_dec   = (const float*)d_in[6];
    const float* W_out   = (const float*)d_in[7];
    const float* b_out   = (const float*)d_in[8];
    float*       out     = (float*)d_out;

    // ---- workspace carve-up (256B aligned) ----
    char* w = (char*)d_ws;
    size_t off = 0;
    auto alloc = [&](size_t bytes) -> void* {
        void* p = w + off;
        off += (bytes + 255) & ~(size_t)255;
        return p;
    };
    unsigned short* frames_bf = (unsigned short*)alloc((size_t)TE * B * DIN * 2);
    unsigned short* WtEx      = (unsigned short*)alloc((size_t)G4 * DIN * 2);  // [2048,2048]
    unsigned short* WtEh      = (unsigned short*)alloc((size_t)G4 * H   * 2);  // [2048,512]
    unsigned short* WtDx      = (unsigned short*)alloc((size_t)G4 * DWP * 2);  // [2048,320]
    unsigned short* WtDh      = (unsigned short*)alloc((size_t)G4 * H   * 2);  // [2048,512]
    unsigned short* WtO       = (unsigned short*)alloc((size_t)NPO * H  * 2);  // [320,512]
    unsigned short* capemb    = (unsigned short*)alloc((size_t)TD * B * DWP * 2);
    float*          XgE       = (float*)alloc((size_t)TE * B * G4 * 4);        // enc gates
    float*          XgD       = (float*)alloc((size_t)TD * B * G4 * 4);        // dec gates
    float*          cstate    = (float*)alloc((size_t)B * H * 4);
    unsigned short* hbf       = (unsigned short*)alloc((size_t)B * H * 2);
    unsigned short* dec_hs    = (unsigned short*)alloc((size_t)TD * B * H * 2);
    float*          proj      = (float*)alloc((size_t)TD * B * NPO * 4);

    const int TPB = 256;

    // 1) convert inputs / weights to bf16 (transposed weights: [N, Kpad])
    conv_frames<<<gsz((long)B * TE * DIN, TPB), TPB, 0, stream>>>(frames, frames_bf, B, TE, DIN);
    conv_weight<<<gsz((long)G4 * DIN, TPB), TPB, 0, stream>>>(W_enc, G4, 0,    DIN, DIN, G4, G4, WtEx);
    conv_weight<<<gsz((long)G4 * H,   TPB), TPB, 0, stream>>>(W_enc, G4, DIN,  H,   H,   G4, G4, WtEh);
    conv_weight<<<gsz((long)G4 * DWP, TPB), TPB, 0, stream>>>(W_dec, G4, 0,    DW,  DWP, G4, G4, WtDx);
    conv_weight<<<gsz((long)G4 * H,   TPB), TPB, 0, stream>>>(W_dec, G4, DW,   H,   H,   G4, G4, WtDh);
    conv_weight<<<gsz((long)NPO * H,  TPB), TPB, 0, stream>>>(W_out, DW, 0,    H,   H,   DW, NPO, WtO);
    gather_emb <<<gsz((long)TD * B * DWP, TPB), TPB, 0, stream>>>(ids, emb, capemb, TD, B, DW, DWP);

    // 2) big time-parallel input-gate GEMMs (bias folded in)
    {   // XgE[T*B, 4H] = frames_bf x WtEx^T + b_enc   (M=8192, N=2048, K=2048)
        unsigned blocks = (unsigned)((TE * B / 128) * (G4 / 64));
        gemm_bf16_wmma<<<blocks, 256, 0, stream>>>(
            frames_bf, DIN, WtEx, DIN, XgE, G4, b_enc, G4, TE * B, G4, DIN, 0);
    }
    {   // XgD[T*B, 4H] = capemb x WtDx^T + b_dec      (M=4096, N=2048, K=320)
        unsigned blocks = (unsigned)((TD * B / 128) * (G4 / 64));
        gemm_bf16_wmma<<<blocks, 256, 0, stream>>>(
            capemb, DWP, WtDx, DWP, XgD, G4, b_dec, G4, TD * B, G4, DWP, 0);
    }

    // 3) zero initial LSTM state
    (void)hipMemsetAsync(cstate, 0, (size_t)B * H * 4, stream);
    (void)hipMemsetAsync(hbf,    0, (size_t)B * H * 2, stream);

    const unsigned recBlocks = (unsigned)((B / 128) * (G4 / 64));   // 32 blocks

    // 4) encoder recurrence: gates[t] += h x W_h ; cell update
    for (int t = 0; t < TE; ++t) {
        float* gates = XgE + (size_t)t * B * G4;
        gemm_bf16_wmma<<<recBlocks, 256, 0, stream>>>(
            hbf, H, WtEh, H, gates, G4, nullptr, 0, B, G4, H, 1);
        lstm_cell<<<gsz((long)B * H, TPB), TPB, 0, stream>>>(
            gates, cstate, hbf, nullptr, B, H);
    }

    // 5) decoder recurrence (teacher-forced), record h_t
    for (int t = 0; t < TD; ++t) {
        float* gates = XgD + (size_t)t * B * G4;
        gemm_bf16_wmma<<<recBlocks, 256, 0, stream>>>(
            hbf, H, WtDh, H, gates, G4, nullptr, 0, B, G4, H, 1);
        lstm_cell<<<gsz((long)B * H, TPB), TPB, 0, stream>>>(
            gates, cstate, hbf, dec_hs + (size_t)t * B * H, B, H);
    }

    // 6) output projection: proj[T*B, NPO] = dec_hs x WtO^T + b_out (M=4096,N=320,K=512)
    {
        unsigned blocks = (unsigned)((TD * B / 128) * (NPO / 64));
        gemm_bf16_wmma<<<blocks, 256, 0, stream>>>(
            dec_hs, H, WtO, H, proj, NPO, b_out, DW, TD * B, NPO, H, 0);
    }

    // 7) strip padding + [T,B] -> [B,T] into d_out
    write_out<<<gsz((long)B * TD * DW, TPB), TPB, 0, stream>>>(proj, out, TD, B, DW, NPO);
}
